// JointMultiHeadedAttention_38465727103378
// MI455X (gfx1250) — matrix operand order repaired and checked by reading, not compiled, not run on hardware
//
#include <hip/hip_runtime.h>
#include <hip/hip_bf16.h>

// ---------------- problem constants ----------------
#define Hh 16
#define DMM 1024
#define Dd 64
#define Bb 2
#define SLL 1024
#define CLL 512
#define TLL (SLL + CLL)   // 1536

typedef __attribute__((ext_vector_type(16))) __bf16 v16bf;
typedef __attribute__((ext_vector_type(8)))  __bf16 v8bf;
typedef __attribute__((ext_vector_type(8)))  float  v8f;
typedef __attribute__((ext_vector_type(4)))  float  v4f;
typedef __attribute__((ext_vector_type(2)))  unsigned int v2u;

union ABu { v16bf v; v8bf h[2]; };

static __device__ __forceinline__ unsigned int f32_to_bf16(float f) {
    union { float f; unsigned u; } x; x.f = f;
    unsigned r = x.u + 0x7FFFu + ((x.u >> 16) & 1u);   // RNE
    return r >> 16;
}

static __device__ __forceinline__ v8f wmma_bf16(const v16bf& a, const v16bf& b, const v8f& c) {
    return __builtin_amdgcn_wmma_f32_16x16x32_bf16(false, a, false, b, (short)0, c, false, false);
}

// ---------------- f32 -> bf16 conversion (vectorized, NT input reads) ----------------
__global__ void cvt_bf16(const float* __restrict__ src, unsigned short* __restrict__ dst, int n4) {
    int i = blockIdx.x * blockDim.x + threadIdx.x;
    int stride = gridDim.x * blockDim.x;
    for (; i < n4; i += stride) {
        v4f s = __builtin_nontemporal_load((const v4f*)src + i);
        v2u p;
        p.x = f32_to_bf16(s.x) | (f32_to_bf16(s.y) << 16);
        p.y = f32_to_bf16(s.z) | (f32_to_bf16(s.w) << 16);
        *((v2u*)dst + i) = p;
    }
}

// ---------------- projection GEMM: Y = X @ W^T + b, scatter to head layout ----------------
// 32x64 output strip per wave: 2 A-frags x 4 B-frags -> 8 accumulators, 8 WMMA / k-step.
// rows_per_batch is a power of two -> passed as rpb_shift.
__global__ void proj_gemm(const unsigned short* __restrict__ X,
                          const unsigned short* __restrict__ W,
                          const float* __restrict__ bias,
                          unsigned short* __restrict__ out,
                          int Mrows, int rpb_shift, int out_len,
                          int tok_base, float scale, int transposed)
{
    const int lane = threadIdx.x & 31;
    const int wave = threadIdx.x >> 5;
    const int gw   = blockIdx.x * (blockDim.x >> 5) + wave;
    const int NT4  = DMM / 64;                   // 16 n-strips
    const int mt = gw / NT4, nt = gw % NT4;
    const int m0 = mt * 32, n0 = nt * 64;
    if (m0 >= Mrows) return;

    const int half = lane >> 4;
    const int l15  = lane & 15;
    const int khA  = half * 8;
    const int kbB  = half * 16;

    const unsigned short* Arow0 = X + (size_t)(m0 + l15) * DMM;
    const unsigned short* Arow1 = X + (size_t)(m0 + 16 + l15) * DMM;
    const unsigned short* Brow[4];
#pragma unroll
    for (int t = 0; t < 4; ++t) Brow[t] = W + (size_t)(n0 + 16 * t + l15) * DMM;

    v8f c[2][4];
#pragma unroll
    for (int g = 0; g < 2; ++g)
#pragma unroll
        for (int t = 0; t < 4; ++t) c[g][t] = (v8f){};

    for (int k0 = 0; k0 < DMM; k0 += 32) {
        ABu aL, aH;
        aL.v = *(const v16bf*)(Arow0 + k0 + kbB);
        aH.v = *(const v16bf*)(Arow1 + k0 + kbB);
#pragma unroll
        for (int t = 0; t < 4; ++t) {
            v16bf b = *(const v16bf*)(Brow[t] + k0 + kbB);
            c[0][t] = wmma_bf16(aL.v, b, c[0][t]);
            c[1][t] = wmma_bf16(aH.v, b, c[1][t]);
        }
    }

    const int rpb_mask = (1 << rpb_shift) - 1;
#pragma unroll
    for (int t = 0; t < 4; ++t) {
        const int n_l = n0 + 16 * t + l15;
        const float bn = bias[n_l];
        const int h = n_l >> 6, d = n_l & 63;
#pragma unroll
        for (int g = 0; g < 2; ++g) {
#pragma unroll
            for (int j = 0; j < 8; ++j) {
                int m = m0 + g * 16 + j + half * 8;
                int bidx = m >> rpb_shift;
                int tk = (m & rpb_mask) + tok_base;
                float y = (c[g][t][j] + bn) * scale;
                size_t idx;
                if (transposed) idx = ((size_t)(bidx * Hh + h) * Dd + d) * (size_t)out_len + tk;
                else            idx = ((size_t)(bidx * Hh + h) * (size_t)out_len + tk) * Dd + d;
                out[idx] = (unsigned short)f32_to_bf16(y);
            }
        }
    }
}

// ---------------- fused attention: scores + softmax + P@V ----------------
// one workgroup (256 thr / 8 waves) per (b, h, 16-query tile)
__global__ void attn_kernel(const unsigned short* __restrict__ Qb,   // [B,H,SL,64]
                            const unsigned short* __restrict__ Kb,   // [B,H,TL,64]
                            const unsigned short* __restrict__ Vt,   // [B,H,64,TL]
                            const unsigned char*  __restrict__ selfm,// [B,SL,SL]
                            const unsigned char*  __restrict__ ctxm, // [B,CL]
                            const float* __restrict__ cbias_p,
                            float* __restrict__ attnOut,             // [B,H,SL,TL]
                            float* __restrict__ topOut,              // [B,SL,CL]
                            unsigned short* __restrict__ CtxB)       // [B*SL, DM] bf16
{
    __shared__ float          S[16][TLL];           // 96 KB  raw scores
    __shared__ unsigned short Pb[16][TLL];          // 48 KB  bf16 probabilities
    __shared__ float          Cacc[16][Dd];         // 4 KB   context accumulator

    const int wg = blockIdx.x;                      // B*H*(SL/16)
    const int qt = wg % (SLL / 16);
    const int bh = wg / (SLL / 16);
    const int b  = bh / Hh, h = bh % Hh;
    const int q0 = qt * 16;
    const int tid  = threadIdx.x;
    const int lane = tid & 31, wave = tid >> 5;
    const int half = lane >> 4, l15 = lane & 15;
    const float cbias = cbias_p[0];

    // Phase-C coordinates (hoisted so Phase A can prefetch V rows)
    const int ntile = wave & 3;
    const int kh    = wave >> 2;
    const int kbeg  = kh * (TLL / 2), kend = kbeg + (TLL / 2);
    const unsigned short* Vrow = Vt + ((size_t)bh * Dd + ntile * 16 + l15) * TLL;

    // ---- Phase A: S = Q K^T (+bias, mask) ----
    const unsigned short* Qrow = Qb + ((size_t)bh * SLL + q0 + l15) * Dd;
    ABu a0, a1;
    a0.v = *(const v16bf*)(Qrow + half * 16);
    a1.v = *(const v16bf*)(Qrow + 32 + half * 16);

    for (int kt = wave; kt < TLL / 16; kt += 8) {
        const int key_l = kt * 16 + l15;
        const unsigned short* Krow = Kb + ((size_t)bh * TLL + key_l) * Dd;
        v16bf b0 = *(const v16bf*)(Krow + half * 16);
        v16bf b1 = *(const v16bf*)(Krow + 32 + half * 16);
        v8f c = {};
        c = wmma_bf16(a0.v, b0, c);
        c = wmma_bf16(a1.v, b1, c);
#pragma unroll
        for (int j = 0; j < 8; ++j) {
            int m = j + half * 8;
            int q = q0 + m;
            float s = c[j];
            if (key_l >= SLL) {
                s += cbias;
                if (ctxm[b * CLL + key_l - SLL]) s = -1e18f;
            } else {
                if (selfm[((size_t)b * SLL + q) * SLL + key_l]) s = -1e18f;
            }
            S[m][key_l] = s;
        }
    }

    // prefetch V cachelines consumed in Phase C (global_prefetch_b8)
    for (int pk = kbeg; pk < kend; pk += 64)
        __builtin_prefetch(Vrow + pk, 0, 1);

    __syncthreads();

    // ---- Phase B: row softmax (16 threads per row, float4 vectorized) ----
    const int row = tid >> 4, cix = tid & 15;
    float mx = -3.0e38f;
    for (int kb = cix * 4; kb < TLL; kb += 64) {
        v4f s = *(const v4f*)&S[row][kb];
        mx = fmaxf(mx, fmaxf(fmaxf(s.x, s.y), fmaxf(s.z, s.w)));
    }
#pragma unroll
    for (int o = 8; o > 0; o >>= 1) mx = fmaxf(mx, __shfl_xor(mx, o, 16));

    float sum = 0.f;
    for (int kb = cix * 4; kb < TLL; kb += 64) {
        v4f s = *(const v4f*)&S[row][kb];
        sum += __expf(s.x - mx) + __expf(s.y - mx) + __expf(s.z - mx) + __expf(s.w - mx);
    }
#pragma unroll
    for (int o = 8; o > 0; o >>= 1) sum += __shfl_xor(sum, o, 16);
    const float inv = 1.0f / sum;

    const int q = q0 + row;
    float* arow = attnOut + ((size_t)bh * SLL + q) * TLL;
    float csum = 0.f;
    for (int kb = cix * 4; kb < TLL; kb += 64) {
        v4f s = *(const v4f*)&S[row][kb];
        v4f p;
        p.x = __expf(s.x - mx) * inv;
        p.y = __expf(s.y - mx) * inv;
        p.z = __expf(s.z - mx) * inv;
        p.w = __expf(s.w - mx) * inv;
        __builtin_nontemporal_store(p, (v4f*)(arow + kb));   // streamed, never re-read
        v2u pk;
        pk.x = f32_to_bf16(p.x) | (f32_to_bf16(p.y) << 16);
        pk.y = f32_to_bf16(p.z) | (f32_to_bf16(p.w) << 16);
        *(v2u*)&Pb[row][kb] = pk;
        if (kb >= SLL) csum += p.x + p.y + p.z + p.w;
    }
    if (h == 0) {
#pragma unroll
        for (int o = 8; o > 0; o >>= 1) csum += __shfl_xor(csum, o, 16);
        const float cinv = 1.0f / csum;
        const float scl = inv * cinv;
        float* trow = topOut + ((size_t)b * SLL + q) * CLL;
        for (int kb = SLL + cix * 4; kb < TLL; kb += 64) {
            v4f s = *(const v4f*)&S[row][kb];
            v4f p;
            p.x = __expf(s.x - mx) * scl;
            p.y = __expf(s.y - mx) * scl;
            p.z = __expf(s.z - mx) * scl;
            p.w = __expf(s.w - mx) * scl;
            __builtin_nontemporal_store(p, (v4f*)(trow + kb - SLL));
        }
    }
    __syncthreads();

    // ---- Phase C: context = P @ V  (waves split 4 n-tiles x 2 K-halves) ----
    v8f acc = {};
    for (int k0 = kbeg; k0 < kend; k0 += 32) {
        ABu a;
        a.v = *(const v16bf*)(&Pb[l15][k0 + half * 16]);
        v16bf bb = *(const v16bf*)(Vrow + k0 + half * 16);
        acc = wmma_bf16(a.v, bb, acc);
    }
    if (wave < 4) {
#pragma unroll
        for (int j = 0; j < 8; ++j) Cacc[j + half * 8][ntile * 16 + l15] = acc[j];
    }
    __syncthreads();
    if (wave >= 4) {
#pragma unroll
        for (int j = 0; j < 8; ++j) Cacc[j + half * 8][ntile * 16 + l15] += acc[j];
    }
    __syncthreads();

#pragma unroll
    for (int e = 0; e < 4; ++e) {
        int flat = tid * 4 + e;        // 0..1023
        int m = flat >> 6, d = flat & 63;
        CtxB[((size_t)b * SLL + q0 + m) * DMM + h * Dd + d] = (unsigned short)f32_to_bf16(Cacc[m][d]);
    }
}

// ---------------- output projection: out = Ctx @ Wo^T + bo (f32 out), 32x64/wave ----------------
__global__ void out_gemm(const unsigned short* __restrict__ X,
                         const unsigned short* __restrict__ W,
                         const float* __restrict__ bias,
                         float* __restrict__ out, int Mrows)
{
    const int lane = threadIdx.x & 31;
    const int wave = threadIdx.x >> 5;
    const int gw   = blockIdx.x * (blockDim.x >> 5) + wave;
    const int NT4  = DMM / 64;
    const int mt = gw / NT4, nt = gw % NT4;
    const int m0 = mt * 32, n0 = nt * 64;
    if (m0 >= Mrows) return;

    const int half = lane >> 4;
    const int l15  = lane & 15;
    const unsigned short* Arow0 = X + (size_t)(m0 + l15) * DMM;
    const unsigned short* Arow1 = X + (size_t)(m0 + 16 + l15) * DMM;
    const unsigned short* Brow[4];
#pragma unroll
    for (int t = 0; t < 4; ++t) Brow[t] = W + (size_t)(n0 + 16 * t + l15) * DMM;

    v8f c[2][4];
#pragma unroll
    for (int g = 0; g < 2; ++g)
#pragma unroll
        for (int t = 0; t < 4; ++t) c[g][t] = (v8f){};

    for (int k0 = 0; k0 < DMM; k0 += 32) {
        ABu aL, aH;
        aL.v = *(const v16bf*)(Arow0 + k0 + half * 16);
        aH.v = *(const v16bf*)(Arow1 + k0 + half * 16);
#pragma unroll
        for (int t = 0; t < 4; ++t) {
            v16bf b = *(const v16bf*)(Brow[t] + k0 + half * 16);
            c[0][t] = wmma_bf16(aL.v, b, c[0][t]);
            c[1][t] = wmma_bf16(aH.v, b, c[1][t]);
        }
    }

#pragma unroll
    for (int t = 0; t < 4; ++t) {
        const int n_l = n0 + 16 * t + l15;
        const float bn = bias[n_l];
#pragma unroll
        for (int g = 0; g < 2; ++g) {
#pragma unroll
            for (int j = 0; j < 8; ++j) {
                int m = m0 + g * 16 + j + half * 8;
                out[(size_t)m * DMM + n_l] = c[g][t][j] + bn;
            }
        }
    }
}

// ---------------- host launch ----------------
extern "C" void kernel_launch(void* const* d_in, const int* in_sizes, int n_in,
                              void* d_out, int out_size, void* d_ws, size_t ws_size,
                              hipStream_t stream) {
    const float* self_kvq = (const float*)d_in[0];
    const float* ctx_kv   = (const float*)d_in[1];
    const unsigned char* self_mask = (const unsigned char*)d_in[2];
    const unsigned char* ctx_mask  = (const unsigned char*)d_in[3];
    const float* Wq  = (const float*)d_in[4];   const float* bq  = (const float*)d_in[5];
    const float* Wks = (const float*)d_in[6];   const float* bks = (const float*)d_in[7];
    const float* Wvs = (const float*)d_in[8];   const float* bvs = (const float*)d_in[9];
    const float* Wkc = (const float*)d_in[10];  const float* bkc = (const float*)d_in[11];
    const float* Wvc = (const float*)d_in[12];  const float* bvc = (const float*)d_in[13];
    const float* ctx_bias = (const float*)d_in[14];
    const float* Wo  = (const float*)d_in[15];  const float* bo  = (const float*)d_in[16];

    float* out_proj = (float*)d_out;                                  // [B,SL,DM]
    float* top_attn = out_proj + (size_t)Bb * SLL * DMM;              // [B,SL,CL]
    float* attn     = top_attn + (size_t)Bb * SLL * CLL;              // [B,H,SL,TL]

    // workspace partition (bf16 = ushort)
    unsigned short* ws = (unsigned short*)d_ws;
    const size_t N_X  = (size_t)Bb * SLL * DMM;      // 2M
    const size_t N_XC = (size_t)Bb * CLL * DMM;      // 1M
    const size_t N_W  = (size_t)DMM * DMM;           // 1M
    const size_t N_Q  = (size_t)Bb * Hh * SLL * Dd;  // 2M
    const size_t N_KV = (size_t)Bb * Hh * TLL * Dd;  // 3M

    unsigned short* Xs  = ws;              // self activations bf16
    unsigned short* Xc  = Xs  + N_X;
    unsigned short* Wqb = Xc  + N_XC;
    unsigned short* Wksb= Wqb + N_W;
    unsigned short* Wvsb= Wksb+ N_W;
    unsigned short* Wkcb= Wvsb+ N_W;
    unsigned short* Wvcb= Wkcb+ N_W;
    unsigned short* Wob = Wvcb+ N_W;
    unsigned short* Qb  = Wob + N_W;       // [B,H,SL,64]
    unsigned short* Kb  = Qb  + N_Q;       // [B,H,TL,64]
    unsigned short* Vt  = Kb  + N_KV;      // [B,H,64,TL]
    unsigned short* Ctx = Vt  + N_KV;      // [B*SL, DM]

    // 1) bf16 conversions (float4-vectorized)
    cvt_bf16<<<2048, 256, 0, stream>>>(self_kvq, Xs,  (int)(N_X  / 4));
    cvt_bf16<<<2048, 256, 0, stream>>>(ctx_kv,   Xc,  (int)(N_XC / 4));
    cvt_bf16<<<2048, 256, 0, stream>>>(Wq,  Wqb, (int)(N_W / 4));
    cvt_bf16<<<2048, 256, 0, stream>>>(Wks, Wksb,(int)(N_W / 4));
    cvt_bf16<<<2048, 256, 0, stream>>>(Wvs, Wvsb,(int)(N_W / 4));
    cvt_bf16<<<2048, 256, 0, stream>>>(Wkc, Wkcb,(int)(N_W / 4));
    cvt_bf16<<<2048, 256, 0, stream>>>(Wvc, Wvcb,(int)(N_W / 4));
    cvt_bf16<<<2048, 256, 0, stream>>>(Wo,  Wob, (int)(N_W / 4));

    // 2) projections: 32x64 strip per wave, 8 waves per 256-thr block
    const int Mself = Bb * SLL;   // 2048 rows
    const int Mctx  = Bb * CLL;   // 1024 rows
    const int blkSelf = (Mself / 32) * (DMM / 64) / 8;  // 128
    const int blkCtx  = (Mctx  / 32) * (DMM / 64) / 8;  // 64
    const float qscale = 0.125f;  // 1/sqrt(64)
    const int shSelf = 10;        // log2(SL)
    const int shCtx  = 9;         // log2(CL)

    proj_gemm<<<blkSelf, 256, 0, stream>>>(Xs, Wqb,  bq,  Qb, Mself, shSelf, SLL, 0,   qscale, 0);
    proj_gemm<<<blkSelf, 256, 0, stream>>>(Xs, Wksb, bks, Kb, Mself, shSelf, TLL, 0,   1.0f,   0);
    proj_gemm<<<blkSelf, 256, 0, stream>>>(Xs, Wvsb, bvs, Vt, Mself, shSelf, TLL, 0,   1.0f,   1);
    proj_gemm<<<blkCtx,  256, 0, stream>>>(Xc, Wkcb, bkc, Kb, Mctx,  shCtx,  TLL, SLL, 1.0f,   0);
    proj_gemm<<<blkCtx,  256, 0, stream>>>(Xc, Wvcb, bvc, Vt, Mctx,  shCtx,  TLL, SLL, 1.0f,   1);

    // 3) fused attention (2048 workgroups: B*H*(SL/16))
    attn_kernel<<<Bb * Hh * (SLL / 16), 256, 0, stream>>>(
        Qb, Kb, Vt, self_mask, ctx_mask, ctx_bias, attn, top_attn, Ctx);

    // 4) output projection
    out_gemm<<<blkSelf, 256, 0, stream>>>(Ctx, Wob, bo, out_proj, Mself);
}
